// DiagonalLSTMCell_81913616270014
// MI455X (gfx1250) — matrix operand order repaired and checked by reading, not compiled, not run on hardware
//
#include <hip/hip_runtime.h>
#include <hip/hip_bf16.h>

// MI455X / gfx1250 DiagonalLSTM.
// Kernel 1: z_is = W_is @ x + b_is  (bf16 WMMA GEMM, 128MB intermediate in d_ws)
// Kernel 2: 127-step diagonal scan, one workgroup per batch (serial chain is the
//           binding constraint): W_ss A-fragments pinned in VGPRs, h-state kept in
//           LDS as pre-packed bf16 (h[c,w-1],h[c,w]) dwords so each WMMA B-fragment
//           is 8 aligned ds_load_b32, c-state f32 in registers, gates via TRANS ops.

typedef __attribute__((ext_vector_type(16))) __bf16 v16bf;
typedef __attribute__((ext_vector_type(8)))  float  v8f;

union BF16x16 { v16bf bf; unsigned int u[8]; };

#define B_  16
#define CIN 64
#define HH  64
#define WW  64
#define HD  128
#define G4  512   // 4*HD
#define CK  256   // HD*K
#define TT  127   // 2W-1

__device__ __forceinline__ unsigned short f2bfbits(float f) {
    __bf16 b = (__bf16)f;
    unsigned short s;
    __builtin_memcpy(&s, &b, 2);
    return s;
}
__device__ __forceinline__ float sigmoid_fast(float x) {
    // v_exp + v_rcp (TRANS ops); exact saturation at the tails
    return __builtin_amdgcn_rcpf(1.0f + __expf(-x));
}
__device__ __forceinline__ float tanh_fast(float x) {
    // tanh(x) = 1 - 2/(exp(2x)+1); exp->inf => 1, exp->0 => -1
    return 1.0f - 2.0f * __builtin_amdgcn_rcpf(__expf(2.0f * x) + 1.0f);
}

// ---------------------------------------------------------------------------
// Kernel 1: z_is[b][o][h][w] = sum_c W_is[o][c] * x[b][c][h][w] + b_is[o]
// Grid (4, 64, 16): x = 128-row o-group, y = h, z = b. Block = 256 (8 waves).
// Each wave: one 16-row M-tile over 64 w-columns, K=64 (2 x K32 WMMA).
// All fragment loads are dword LDS loads (K-contiguous layouts).
// ---------------------------------------------------------------------------
__global__ __launch_bounds__(256)
void zis_gemm_kernel(const float* __restrict__ x, const float* __restrict__ Wis,
                     const float* __restrict__ bis, float* __restrict__ zis)
{
    __shared__ unsigned short xsT[WW][CIN + 4];   // [w][c] bf16 bits (transposed)
    __shared__ unsigned short ws[128][CIN + 2];   // [o_local][c] bf16 bits

    const int b   = blockIdx.z;
    const int h   = blockIdx.y;
    const int mg  = blockIdx.x;                   // 0..3 -> rows [mg*128, +128)
    const int tid = threadIdx.x;

    for (int i = tid; i < CIN * WW; i += 256) {
        int c = i >> 6, w = i & 63;
        xsT[w][c] = f2bfbits(x[(((b * CIN + c) * HH + h) << 6) + w]);
    }
    for (int i = tid; i < 128 * CIN; i += 256) {
        int ol = i >> 6, c = i & 63;
        ws[ol][c] = f2bfbits(Wis[((mg * 128 + ol) << 6) + c]);
    }
    __syncthreads();

    const int lane  = tid & 31;
    const int wv    = tid >> 5;                   // 0..7 -> M-tile
    const int n     = lane & 15;                  // N col within tile / A row m
    const int hf    = lane >> 4;                  // lane half
    const int obase = wv * 16;

    for (int nt = 0; nt < 4; ++nt) {
        v8f acc = {0.f, 0.f, 0.f, 0.f, 0.f, 0.f, 0.f, 0.f};
        const int ncol = nt * 16 + n;
        #pragma unroll
        for (int kb = 0; kb < 2; ++kb) {
            BF16x16 A, Bf;
            // A 16x32: elems 0-7 <-> K=8*half+e, 8-15 <-> K=16+8*half+e (pairs = dwords)
            #pragma unroll
            for (int j = 0; j < 4; ++j) {
                A.u[j]     = *(const unsigned int*)&ws[obase + n][kb * 32 + 8 * hf + 2 * j];
                A.u[j + 4] = *(const unsigned int*)&ws[obase + n][kb * 32 + 16 + 8 * hf + 2 * j];
            }
            // B 32x16: elems 0-7 <-> K=8*half+e, 8-15 <-> K=16+8*half+e (pairs = dwords)
            #pragma unroll
            for (int j = 0; j < 8; ++j)
                Bf.u[j] = *(const unsigned int*)&xsT[ncol][kb * 32 + 8 * hf + 2 * (j & 3) + 16 * (j >> 2)];
            acc = __builtin_amdgcn_wmma_f32_16x16x32_bf16(
                      false, A.bf, false, Bf.bf, (short)0, acc, false, false);
        }
        #pragma unroll
        for (int v = 0; v < 8; ++v) {
            int o = mg * 128 + obase + v + 8 * hf;   // D: M = v + 8*half
            zis[(((b * G4 + o) * HH + h) << 6) + ncol] = acc[v] + bis[o];
        }
    }
}

// ---------------------------------------------------------------------------
// Kernel 2: serial diagonal scan. Grid = 16 (one block per batch),
// block = 1024 threads = 32 waves. Wave w owns gate rows [16w, 16w+16).
// h-state LDS layout: hp[c][2*w+0] = bf16 h[c][w-1], hp[c][2*w+1] = bf16 h[c][w]
// -> the dword at pair w is exactly one packed B-fragment element (ck = 2c+k).
// ---------------------------------------------------------------------------
__global__ __launch_bounds__(1024)
void diag_lstm_scan_kernel(const float* __restrict__ zis,
                           const float* __restrict__ h0,
                           const float* __restrict__ c0,
                           const float* __restrict__ Wss,
                           const float* __restrict__ bss,
                           float* __restrict__ out)
{
    __shared__ unsigned short hp[HD][132];   // pairs w=0..64 (+pad); dword aligned rows
    __shared__ float          act[G4][65];   // activated gates, [o][h]

    const int b    = blockIdx.x;
    const int tid  = threadIdx.x;
    const int lane = tid & 31;
    const int wv   = tid >> 5;               // 0..31 -> 16-row o-tile
    const int n    = lane & 15;
    const int hf   = lane >> 4;
    const int obase = wv * 16;

    // ---- init packed hidden state ----
    for (int i = tid; i < HD * HH; i += 1024) {
        int c = i >> 6, h = i & 63;
        unsigned short bits = f2bfbits(h0[(b * HD + c) * HH + h]);
        hp[c][2 * h + 1]   = bits;           // hi of pair h   : h[c][h]
        hp[c][2 * (h + 1)] = bits;           // lo of pair h+1 : h[c][(h+1)-1]
    }
    if (tid < HD) hp[tid][0] = 0;            // h[c][-1] = 0

    // ---- cell state registers: thread -> (hd = tid>>3, h in [(tid&7)*8, +8)) ----
    const int u_hd = tid >> 3;
    const int u_h0 = (tid & 7) * 8;
    float creg[8];
    #pragma unroll
    for (int j = 0; j < 8; ++j)
        creg[j] = c0[(b * HD + u_hd) * HH + u_h0 + j];

    // ---- pin A fragments (W_ss rows for this wave) in VGPRs for all 127 steps ----
    v16bf afrag[8];
    #pragma unroll
    for (int kb = 0; kb < 8; ++kb) {
        #pragma unroll
        for (int e = 0; e < 8; ++e) {
            afrag[kb][e]     = (__bf16)Wss[(obase + n) * CK + kb * 32 + 16 * hf + e];
            afrag[kb][e + 8] = (__bf16)Wss[(obase + n) * CK + kb * 32 + 16 * hf + 8 + e];
        }
    }
    float biasr[8];
    #pragma unroll
    for (int v = 0; v < 8; ++v) biasr[v] = bss[obase + v + 8 * hf];
    const bool is_g = (obase >= 3 * HD);     // rows 384..511 -> tanh, else sigmoid

    __syncthreads();

    for (int t = 0; t < TT; ++t) {
        // ---- phase 1: recurrent GEMM (8 dword DS loads per B-fragment) ----
        for (int nt = 0; nt < 4; ++nt) {
            const int hcol = nt * 16 + n;
            v8f acc = {0.f, 0.f, 0.f, 0.f, 0.f, 0.f, 0.f, 0.f};
            #pragma unroll
            for (int kb = 0; kb < 8; ++kb) {
                BF16x16 Bf;
                #pragma unroll
                for (int j = 0; j < 8; ++j) {
                    int c = kb * 16 + 8 * hf + j;     // elems (2j,2j+1) <-> (h[c,hcol-1],h[c,hcol])
                    Bf.u[j] = *(const unsigned int*)&hp[c][2 * hcol];
                }
                acc = __builtin_amdgcn_wmma_f32_16x16x32_bf16(
                          false, afrag[kb], false, Bf.bf, (short)0, acc, false, false);
            }
            const int  w_idx = t - hcol;              // skew: z_steps[t][.,h] = z_is[.,h,t-h]
            const bool inw   = (w_idx >= 0) & (w_idx < WW);
            #pragma unroll
            for (int v = 0; v < 8; ++v) {
                int   o = obase + v + 8 * hf;
                float z = acc[v] + biasr[v];
                if (inw) z += zis[(((b * G4 + o) * HH + hcol) << 6) + w_idx];
                act[o][hcol] = is_g ? tanh_fast(z) : sigmoid_fast(z);
            }
        }
        __syncthreads();

        // ---- phase 2: LSTM cell update; packed dword h writes ----
        unsigned short bits[8];
        #pragma unroll
        for (int j = 0; j < 8; ++j) {
            const int h  = u_h0 + j;
            float iv = act[u_hd         ][h];
            float fv = act[HD   + u_hd  ][h];
            float ov = act[2*HD + u_hd  ][h];
            float gv = act[3*HD + u_hd  ][h];
            creg[j] = fv * creg[j] + iv * gv;
            float hn = ov * tanh_fast(creg[j]);
            bits[j] = f2bfbits(hn);
            int w_idx = t - h;                        // out[b,hd,h,w] = h at t = h+w
            if (w_idx >= 0 && w_idx < WW)
                out[(((b * HD + u_hd) * HH + h) << 6) + w_idx] = hn;
        }
        // interior pairs p = u_h0+1..u_h0+7 fully owned -> dword stores
        #pragma unroll
        for (int j = 1; j < 8; ++j) {
            unsigned int pk = (unsigned int)bits[j - 1] | ((unsigned int)bits[j] << 16);
            *(unsigned int*)&hp[u_hd][2 * (u_h0 + j)] = pk;
        }
        hp[u_hd][2 * u_h0 + 1]       = bits[0];       // hi of boundary pair u_h0
        hp[u_hd][2 * (u_h0 + 8)]     = bits[7];       // lo of boundary pair u_h0+8
        __syncthreads();
    }
}

// ---------------------------------------------------------------------------
extern "C" void kernel_launch(void* const* d_in, const int* in_sizes, int n_in,
                              void* d_out, int out_size, void* d_ws, size_t ws_size,
                              hipStream_t stream) {
    const float* x   = (const float*)d_in[0];
    const float* h0  = (const float*)d_in[1];
    const float* c0  = (const float*)d_in[2];
    const float* Wis = (const float*)d_in[3];
    const float* bis = (const float*)d_in[4];
    const float* Wss = (const float*)d_in[5];
    const float* bss = (const float*)d_in[6];
    float*       out = (float*)d_out;
    float*       zis = (float*)d_ws;   // 16*512*64*64 f32 = 128 MB scratch

    dim3 g1(4, HH, B_);
    zis_gemm_kernel<<<g1, 256, 0, stream>>>(x, Wis, bis, zis);
    diag_lstm_scan_kernel<<<B_, 1024, 0, stream>>>(zis, h0, c0, Wss, bss, out);
}
